// RNN_45990509805909
// MI455X (gfx1250) — matrix operand order repaired and checked by reading, not compiled, not run on hardware
//
#include <hip/hip_runtime.h>

// Problem constants (from reference): T=4096, B=64, D=256, H=256
#define T_STEPS 4096
#define B_DIM   64
#define D_DIM   256
#define H_DIM   256
#define BH      (B_DIM * H_DIM)   // 16384

typedef __bf16 bf16_t;
typedef __attribute__((ext_vector_type(16))) __bf16 v16bf;
typedef __attribute__((ext_vector_type(8)))  __bf16 v8bf;
typedef __attribute__((ext_vector_type(4)))  __bf16 v4bf;
typedef __attribute__((ext_vector_type(8)))  float  v8f;
typedef __attribute__((ext_vector_type(4)))  float  v4f;

// Types for the async-copy builtin: param0 = AS1 (global) int4*, param1 = AS3 (LDS)
typedef __attribute__((ext_vector_type(4))) int i4;
typedef __attribute__((address_space(1))) i4 gi4;
typedef __attribute__((address_space(3))) i4 li4;

// Flat->LDS offset: ISA 10.2 — LDS aperture flat addresses truncate to the
// 32-bit LDS byte offset, so an integer round-trip yields a valid AS3 pointer.
__device__ __forceinline__ li4* to_lds(void* p) {
    return (li4*)(unsigned int)(unsigned long long)p;
}
__device__ __forceinline__ gi4* to_glb(const void* p) {
    return (gi4*)(unsigned long long)p;
}

__device__ __forceinline__ v16bf join16(v8bf lo, v8bf hi) {
    return __builtin_shufflevector(lo, hi, 0,1,2,3,4,5,6,7,8,9,10,11,12,13,14,15);
}

__device__ __forceinline__ v8f wmma_bf16(v16bf a, v16bf b, v8f c) {
    // D = A(16x32 bf16) * B(32x16 bf16) + C(16x16 f32)
    return __builtin_amdgcn_wmma_f32_16x16x32_bf16(
        /*neg_a=*/false, a, /*neg_b=*/false, b,
        /*c_mod=*/(short)0, c, /*reuse_a=*/false, /*reuse_b=*/false);
}

// B fragment (32x16, bf16) for tile (nTile,kTile) from WT stored transposed
// row-major [N][K] with leading dim ld. Contiguous-k layout, same as load_afrag_lds:
//   lane L: n = L%16, kbase = 16*(L/16); VGPR v holds K = kbase+2v, kbase+2v+1
__device__ __forceinline__ v16bf load_bfrag(const bf16_t* __restrict__ WT,
                                            int nTile, int kTile, int lane, int ld) {
    const bf16_t* p = WT + (size_t)(nTile * 16 + (lane & 15)) * ld
                         + kTile * 32 + ((lane >> 4) * 16);
    v8bf lo = *(const v8bf*)p;
    v8bf hi = *(const v8bf*)(p + 8);
    return join16(lo, hi);
}

// A fragment (16x32, bf16) from LDS row-major [M][K]:
//   lane L: m = L%16, g = L/16; VGPR0-7: K = g*16..g*16+15
__device__ __forceinline__ v16bf load_afrag_lds(const bf16_t* At,
                                                int mTile, int kTile, int lane, int ld) {
    const int m  = mTile * 16 + (lane & 15);
    const int kc = kTile * 32 + ((lane >> 4) * 16);
    v8bf lo = *(const v8bf*)(At + m * ld + kc);
    v8bf hi = *(const v8bf*)(At + m * ld + kc + 8);
    return join16(lo, hi);
}

// ---------------------------------------------------------------------------
// Kernel A: transpose+convert weights to bf16 in workspace, fuse biases.
// ---------------------------------------------------------------------------
__global__ void prep_weights(const float* __restrict__ Wih, const float* __restrict__ Whh,
                             const float* __restrict__ bih, const float* __restrict__ bhh,
                             bf16_t* __restrict__ WihT, bf16_t* __restrict__ WhhT,
                             float* __restrict__ bias) {
    const int n = blockIdx.x;
    const int k = threadIdx.x;
    WihT[(size_t)n * D_DIM + k] = (bf16_t)Wih[(size_t)k * H_DIM + n];
    WhhT[(size_t)n * H_DIM + k] = (bf16_t)Whh[(size_t)k * H_DIM + n];
    if (n == 0) bias[k] = bih[k] + bhh[k];
}

// ---------------------------------------------------------------------------
// Kernel B: pre = bf16(xs) @ bf16(Wih) + (bih+bhh), fp32 out (memory-bound).
// M = T*B = 262144, K = 256, N = 256. 4096 blocks x 512 threads.
// ---------------------------------------------------------------------------
__global__ __launch_bounds__(512, 1)
void gemm_pre(const float* __restrict__ xs, const bf16_t* __restrict__ WihT,
              const float* __restrict__ bias, float* __restrict__ pre) {
    extern __shared__ char smem[];
    bf16_t* At = (bf16_t*)smem;  // [64][256] bf16 = 32 KB

    const int tid  = threadIdx.x;
    const int lane = tid & 31;
    const int wave = tid >> 5;        // 0..15 == nTile
    const int g    = lane >> 4;
    const size_t rowBase = (size_t)blockIdx.x * 64;
    const float* xblk = xs + rowBase * (size_t)D_DIM;

    #pragma unroll
    for (int i = 0; i < 8; ++i) {
        int q = tid + i * 512;
        v4f f = ((const v4f*)xblk)[q];
        v4bf b;
        b[0] = (bf16_t)f[0]; b[1] = (bf16_t)f[1];
        b[2] = (bf16_t)f[2]; b[3] = (bf16_t)f[3];
        *((v4bf*)(At + (q >> 6) * D_DIM + (q & 63) * 4)) = b;
    }

    v16bf bfrag[8];
    #pragma unroll
    for (int kt = 0; kt < 8; ++kt) bfrag[kt] = load_bfrag(WihT, wave, kt, lane, D_DIM);

    __syncthreads();

    const float bn = bias[wave * 16 + (lane & 15)];
    const int   n  = wave * 16 + (lane & 15);

    #pragma unroll
    for (int mt = 0; mt < 4; ++mt) {
        v8f acc = {0.f, 0.f, 0.f, 0.f, 0.f, 0.f, 0.f, 0.f};
        #pragma unroll
        for (int kt = 0; kt < 8; ++kt) {
            v16bf a = load_afrag_lds(At, mt, kt, lane, D_DIM);
            acc = wmma_bf16(a, bfrag[kt], acc);
        }
        const size_t m0 = rowBase + mt * 16 + 8 * g;
        #pragma unroll
        for (int r = 0; r < 8; ++r) {
            pre[(m0 + r) * H_DIM + n] = acc[r] + bn;
        }
    }
}

// ---------------------------------------------------------------------------
// Kernel C: sequential recurrence, ONE persistent workgroup (16 waves).
//   h_t = tanh(pre[t] + h_{t-1} @ Whh);  outs[t] = h_t
// Whh pinned in VGPRs. h double-buffered bf16 in LDS (64 KB). pre[t] staged
// via GLOBAL_LOAD_ASYNC_TO_LDS_B128 into a TRIPLE-buffered fp32 LDS region
// (3 x 64 KB), overlapping the next step's memory with this step's WMMAs.
// One s_barrier per step. LDS total = 256 KB of the 320 KB WGP pool.
// ---------------------------------------------------------------------------
__global__ __launch_bounds__(512, 1)
void rnn_recur(const float* __restrict__ init, const bf16_t* __restrict__ WhhT,
               float* __restrict__ out) {
    extern __shared__ char smem[];
    bf16_t* h0   = (bf16_t*)smem;           // [64][256] bf16
    bf16_t* h1   = h0 + BH;
    float*  pbuf = (float*)(smem + 2 * BH * sizeof(bf16_t));  // 3 x [64][256] f32

    const int tid  = threadIdx.x;
    const int lane = tid & 31;
    const int wave = tid >> 5;      // 0..15 == nTile
    const int g    = lane >> 4;
    const int n    = wave * 16 + (lane & 15);

    float* outs = out + BH;

    // Stage h_{-1} = init (fp32) -> LDS bf16 buffer 0
    #pragma unroll
    for (int i = 0; i < 8; ++i) {
        int q = tid + i * 512;
        v4f f = ((const v4f*)init)[q];
        v4bf b;
        b[0] = (bf16_t)f[0]; b[1] = (bf16_t)f[1];
        b[2] = (bf16_t)f[2]; b[3] = (bf16_t)f[3];
        *((v4bf*)(h0 + q * 4)) = b;
    }

    // Pin Whh column block in registers (8 frags x 8 VGPRs = 64 VGPRs)
    v16bf bfrag[8];
    #pragma unroll
    for (int kt = 0; kt < 8; ++kt) bfrag[kt] = load_bfrag(WhhT, wave, kt, lane, H_DIM);

    // Kick off async staging of pre[0] into pbuf slot 0 (ASYNCcnt += 8)
    #pragma unroll
    for (int i = 0; i < 8; ++i) {
        int q = tid + i * 512;  // 16-byte chunk index within the 64 KB tile
        __builtin_amdgcn_global_load_async_to_lds_b128(
            to_glb(outs + q * 4), to_lds(pbuf + q * 4), 0, 0);
    }

    for (int t = 0; t < T_STEPS; ++t) {
        bf16_t* cur  = (t & 1) ? h1 : h0;
        bf16_t* nxt  = (t & 1) ? h0 : h1;
        float*  pcur = pbuf + (t % 3) * BH;

        if (t + 1 < T_STEPS) {
            // Issue async copies for pre[t+1] into slot (t+1)%3, then wait for
            // slot t%3 (in-order completion: <=8 outstanding == only batch t+1)
            float* pnxt = pbuf + ((t + 1) % 3) * BH;
            const float* gsrc = outs + (size_t)(t + 1) * BH;
            #pragma unroll
            for (int i = 0; i < 8; ++i) {
                int q = tid + i * 512;
                __builtin_amdgcn_global_load_async_to_lds_b128(
                    to_glb(gsrc + q * 4), to_lds(pnxt + q * 4), 0, 0);
            }
            asm volatile("s_wait_asynccnt 0x8" ::: "memory");
        } else {
            asm volatile("s_wait_asynccnt 0x0" ::: "memory");
        }
        __syncthreads();  // batch-t staging + h writes from t-1 visible to all

        #pragma unroll
        for (int mt = 0; mt < 4; ++mt) {
            v8f acc = {0.f, 0.f, 0.f, 0.f, 0.f, 0.f, 0.f, 0.f};
            #pragma unroll
            for (int kt = 0; kt < 8; ++kt) {
                v16bf a = load_afrag_lds(cur, mt, kt, lane, H_DIM);
                acc = wmma_bf16(a, bfrag[kt], acc);
            }
            const int m0 = mt * 16 + 8 * g;
            float* outT = outs + (size_t)t * BH;
            #pragma unroll
            for (int r = 0; r < 8; ++r) {
                float v = tanhf(acc[r] + pcur[(m0 + r) * H_DIM + n]);
                outT[(m0 + r) * H_DIM + n] = v;          // outs[t] = h_t (fp32)
                nxt[(m0 + r) * H_DIM + n]  = (bf16_t)v;  // feed next step (bf16)
            }
        }
    }

    // carry = h_{T-1}: copy outs[T-1] -> out[0:BH)
    __threadfence();
    __syncthreads();
    const float* last = outs + (size_t)(T_STEPS - 1) * BH;
    #pragma unroll
    for (int i = 0; i < 8; ++i) {
        int q = tid + i * 512;
        ((v4f*)out)[q] = ((const v4f*)last)[q];
    }
}

// ---------------------------------------------------------------------------
extern "C" void kernel_launch(void* const* d_in, const int* in_sizes, int n_in,
                              void* d_out, int out_size, void* d_ws, size_t ws_size,
                              hipStream_t stream) {
    const float* init = (const float*)d_in[0];
    const float* xs   = (const float*)d_in[1];
    const float* Wih  = (const float*)d_in[2];
    const float* bih  = (const float*)d_in[3];
    const float* Whh  = (const float*)d_in[4];
    const float* bhh  = (const float*)d_in[5];
    float* out = (float*)d_out;                  // [carry(16384) | outs(T*B*H)]

    char* ws = (char*)d_ws;
    bf16_t* WihT = (bf16_t*)ws;                  // 128 KB
    bf16_t* WhhT = (bf16_t*)(ws + 131072);       // 128 KB
    float*  bias = (float*)(ws + 262144);        // 1 KB

    (void)in_sizes; (void)n_in; (void)out_size; (void)ws_size;

    prep_weights<<<H_DIM, 256, 0, stream>>>(Wih, Whh, bih, bhh, WihT, WhhT, bias);

    gemm_pre<<<(T_STEPS * B_DIM) / 64, 512, 64 * D_DIM * sizeof(bf16_t), stream>>>(
        xs, WihT, bias, out + BH);

    // LDS: 2*BH bf16 (h) + 3*BH f32 (pre triple buffer) = 256 KB
    rnn_recur<<<1, 512, 2 * BH * sizeof(bf16_t) + 3 * BH * sizeof(float), stream>>>(
        init, WhhT, out);
}
